// MultiheadedAttention_26027501814630
// MI455X (gfx1250) — matrix operand order repaired and checked by reading, not compiled, not run on hardware
//
#include <hip/hip_runtime.h>
#include <hip/hip_bf16.h>
#include <stdint.h>

// MultiheadedAttention for MI455X (gfx1250, wave32, WMMA f16 16x16x32, f32 acc)
// B=2 S=2048 E=512 H=8 D=64

#define B_   2
#define S_   2048
#define E_   512
#define H_   8
#define D_   64
#define HD_  512
#define NEGV (-1e10f)
#define SCALE_ 0.125f   // 1/sqrt(64)

typedef __attribute__((ext_vector_type(8)))  _Float16 v8h;
typedef __attribute__((ext_vector_type(16))) _Float16 v16h;
typedef __attribute__((ext_vector_type(8)))  float    v8f;
typedef __attribute__((ext_vector_type(4)))  float    f4;

// A-matrix per-lane gather (16x32 f16): lane m = lane&15, K run of 16 at
// k0 = 16*(lane>>4). One contiguous 32B region.
__device__ __forceinline__ v16h load_a16(const _Float16* p) {
  return *(const v16h*)(p);
}
// B-matrix per-lane gather (32x16 f16): lane n = lane&15, K run of 16 at
// k0 = 16*(lane>>4). One contiguous 32B region.
__device__ __forceinline__ v16h load_b16(const _Float16* p) {
  return *(const v16h*)(p);
}
__device__ __forceinline__ v8f wmma16(v16h a, v16h b, v8f c) {
  return __builtin_amdgcn_wmma_f32_16x16x32_f16(false, a, false, b,
                                                (short)0, c, false, false);
}

// Load 16 contiguous f32 (4 x global_load_b128 in flight), convert, and
// store contiguously into LDS as 2 x 16B ds stores.
__device__ __forceinline__ void stage16(_Float16* dst, const float* src) {
  f4 a = *(const f4*)(src +  0);
  f4 b = *(const f4*)(src +  4);
  f4 c = *(const f4*)(src +  8);
  f4 d = *(const f4*)(src + 12);
  v8h h0 = {(_Float16)a.x,(_Float16)a.y,(_Float16)a.z,(_Float16)a.w,
            (_Float16)b.x,(_Float16)b.y,(_Float16)b.z,(_Float16)b.w};
  v8h h1 = {(_Float16)c.x,(_Float16)c.y,(_Float16)c.z,(_Float16)c.w,
            (_Float16)d.x,(_Float16)d.y,(_Float16)d.z,(_Float16)d.w};
  *(v8h*)(dst)     = h0;
  *(v8h*)(dst + 8) = h1;
}
// Load 16 contiguous f32, convert, scatter into LDS with element stride
// (transposed tiles): coalesced global reads, strided ds_store_b16 writes.
__device__ __forceinline__ void stage16_t(_Float16* dst, int stride,
                                          const float* src) {
  f4 a = *(const f4*)(src +  0);
  f4 b = *(const f4*)(src +  4);
  f4 c = *(const f4*)(src +  8);
  f4 d = *(const f4*)(src + 12);
  _Float16 t[16] = {(_Float16)a.x,(_Float16)a.y,(_Float16)a.z,(_Float16)a.w,
                    (_Float16)b.x,(_Float16)b.y,(_Float16)b.z,(_Float16)b.w,
                    (_Float16)c.x,(_Float16)c.y,(_Float16)c.z,(_Float16)c.w,
                    (_Float16)d.x,(_Float16)d.y,(_Float16)d.z,(_Float16)d.w};
  #pragma unroll
  for (int i = 0; i < 16; ++i) dst[i*stride] = t[i];
}
// 8-element variant for the Wq tile (transposed scatter).
__device__ __forceinline__ void stage8_t(_Float16* dst, int stride,
                                         const float* src) {
  f4 a = *(const f4*)(src + 0);
  f4 b = *(const f4*)(src + 4);
  _Float16 t[8] = {(_Float16)a.x,(_Float16)a.y,(_Float16)a.z,(_Float16)a.w,
                   (_Float16)b.x,(_Float16)b.y,(_Float16)b.z,(_Float16)b.w};
  #pragma unroll
  for (int i = 0; i < 8; ++i) dst[i*stride] = t[i];
}

// ---------------------------------------------------------------------------
// Kernel 1: fused Q-projection + flash attention.
// grid = (S/128, B*H), block = 256 threads (8 waves). Wave w owns query rows
// [blockIdx.x*128 + w*16, +16). Output: attn workspace (B,S,H,D) f32.
// LDS: Qs 16K + Ks 8K + Vts 8K + Ps 16K + As0 8K + Bs0 4K = 60 KB
// ---------------------------------------------------------------------------
__global__ __launch_bounds__(256)
void attn_fused_kernel(const float* __restrict__ X,   // (B,S,E)
                       const float* __restrict__ Kg,  // (B,S,H,D)
                       const float* __restrict__ Vg,  // (B,S,H,D)
                       const float* __restrict__ Wq,  // (E, H*D)
                       const uint8_t* __restrict__ mask, // (B,S,S) bool
                       float* __restrict__ attn)      // (B,S,H,D)
{
  __shared__ __align__(32) _Float16 Qs[128*64];    // [qrow][d], pre-scaled
  __shared__ __align__(32) _Float16 Ks[64*64];     // [key][d]
  __shared__ __align__(32) _Float16 Vts[64*64];    // [d][key] (transposed)
  __shared__ __align__(32) _Float16 Ps[8][16*64];  // per-wave P relayout buf
  __shared__ __align__(32) _Float16 As0[128*32];   // phase0 X tile [row][k]
  __shared__ __align__(32) _Float16 Bs0[64*32];    // phase0 Wq tile [n][k]

  const int q0   = blockIdx.x * 128;
  const int bh   = blockIdx.y;
  const int b    = bh >> 3, h = bh & 7;
  const int tid  = threadIdx.x;
  const int wave = tid >> 5, lane = tid & 31;
  const int hi   = lane >> 4, nl = lane & 15;

  const v8f zero8 = {0.f,0.f,0.f,0.f,0.f,0.f,0.f,0.f};

  // ---- Phase 0: Q tile = X(128x512) @ Wq[:, h*64 : h*64+64]  (WMMA) ----
  v8f qacc[4];
  #pragma unroll
  for (int nt = 0; nt < 4; ++nt) qacc[nt] = zero8;

  // staging index precompute
  const int xrow  = tid >> 1, xkoff = (tid & 1) * 16;   // 128x32 X tile
  const int wk    = tid >> 3, wnoff = (tid & 7) * 8;    // 32x64  Wq tile
  const int skey  = tid >> 2, sdoff = (tid & 3) * 16;   // 64x64  K/V tiles

  for (int kk = 0; kk < E_/32; ++kk) {
    __syncthreads();
    // X tile: each thread stages 16 contiguous k (4 x b128 loads, 1 wait)
    stage16(&As0[xrow*32 + xkoff],
            &X[(b*S_ + q0 + xrow)*E_ + kk*32 + xkoff]);
    // Wq tile: coalesced along n, scattered into [n][k] LDS layout
    stage8_t(&Bs0[wnoff*32 + wk], 32,
             &Wq[(kk*32 + wk)*HD_ + h*64 + wnoff]);
    __syncthreads();
    v16h a = load_a16(&As0[(wave*16 + nl)*32 + hi*16]);
    #pragma unroll
    for (int nt = 0; nt < 4; ++nt) {
      v16h bm = load_b16(&Bs0[(nt*16 + nl)*32 + hi*16]);
      qacc[nt] = wmma16(a, bm, qacc[nt]);
    }
  }
  // D-layout -> LDS, fold in 1/sqrt(D)
  #pragma unroll
  for (int nt = 0; nt < 4; ++nt)
    #pragma unroll
    for (int r = 0; r < 8; ++r)
      Qs[(wave*16 + r + 8*hi)*64 + nt*16 + nl] = (_Float16)(qacc[nt][r] * SCALE_);

  // ---- Phase 1: flash-attention over key blocks of 64 ----
  v8f oacc[4];
  #pragma unroll
  for (int ot = 0; ot < 4; ++ot) oacc[ot] = zero8;
  float rmax[8], rsum[8];
  #pragma unroll
  for (int r = 0; r < 8; ++r) { rmax[r] = -3.0e38f; rsum[r] = 0.f; }

  for (int j = 0; j < S_/64; ++j) {
    __syncthreads();
    {
      int gbase = ((b*S_ + j*64 + skey)*H_ + h)*D_ + sdoff;
      stage16(&Ks[skey*64 + sdoff], &Kg[gbase]);         // [key][d]
      stage16_t(&Vts[sdoff*64 + skey], 64, &Vg[gbase]);  // [d][key]
    }
    __syncthreads();

    // S strip (16 rows x 64 keys) = Qs(16x64) @ Ks^T
    v8f sacc[4];
    #pragma unroll
    for (int ct = 0; ct < 4; ++ct) sacc[ct] = zero8;
    #pragma unroll
    for (int kk = 0; kk < 2; ++kk) {
      v16h a = load_a16(&Qs[(wave*16 + nl)*64 + kk*32 + hi*16]);
      #pragma unroll
      for (int ct = 0; ct < 4; ++ct) {
        v16h bm = load_b16(&Ks[(ct*16 + nl)*64 + kk*32 + hi*16]);
        sacc[ct] = wmma16(a, bm, sacc[ct]);
      }
    }

    // mask + block row-max (rows per lane: r + 8*hi)
    float bmax[8];
    #pragma unroll
    for (int r = 0; r < 8; ++r) bmax[r] = -3.0e38f;
    #pragma unroll
    for (int ct = 0; ct < 4; ++ct) {
      #pragma unroll
      for (int r = 0; r < 8; ++r) {
        int q   = q0 + wave*16 + r + 8*hi;
        int key = j*64 + ct*16 + nl;
        float sc = sacc[ct][r];
        if (mask[(b*S_ + q)*S_ + key]) sc = NEGV;
        sacc[ct][r] = sc;
        bmax[r] = fmaxf(bmax[r], sc);
      }
    }
    #pragma unroll
    for (int m = 1; m < 16; m <<= 1)
      #pragma unroll
      for (int r = 0; r < 8; ++r)
        bmax[r] = fmaxf(bmax[r], __shfl_xor(bmax[r], m, 32));

    // online-softmax state update
    float corr[8];
    #pragma unroll
    for (int r = 0; r < 8; ++r) {
      float nm = fmaxf(rmax[r], bmax[r]);
      corr[r]  = __expf(rmax[r] - nm);
      rmax[r]  = nm;
    }
    float bsum[8];
    #pragma unroll
    for (int r = 0; r < 8; ++r) bsum[r] = 0.f;
    #pragma unroll
    for (int ct = 0; ct < 4; ++ct)
      #pragma unroll
      for (int r = 0; r < 8; ++r) {
        float p = __expf(sacc[ct][r] - rmax[r]);
        bsum[r] += p;
        // D-layout -> A-layout relayout via per-wave LDS tile
        Ps[wave][(r + 8*hi)*64 + ct*16 + nl] = (_Float16)p;
      }
    #pragma unroll
    for (int m = 1; m < 16; m <<= 1)
      #pragma unroll
      for (int r = 0; r < 8; ++r)
        bsum[r] += __shfl_xor(bsum[r], m, 32);
    #pragma unroll
    for (int r = 0; r < 8; ++r) rsum[r] = rsum[r]*corr[r] + bsum[r];

    // rescale running O, then O += P @ V
    #pragma unroll
    for (int ot = 0; ot < 4; ++ot)
      #pragma unroll
      for (int r = 0; r < 8; ++r)
        oacc[ot][r] *= corr[r];

    #pragma unroll
    for (int kk = 0; kk < 2; ++kk) {
      v16h a = load_a16(&Ps[wave][nl*64 + kk*32 + hi*16]);
      #pragma unroll
      for (int ot = 0; ot < 4; ++ot) {
        v16h bm = load_b16(&Vts[(ot*16 + nl)*64 + kk*32 + hi*16]);
        oacc[ot] = wmma16(a, bm, oacc[ot]);
      }
    }
  }

  // epilogue: normalize and write attn workspace (B,S,H,D)
  #pragma unroll
  for (int ot = 0; ot < 4; ++ot)
    #pragma unroll
    for (int r = 0; r < 8; ++r) {
      int q = q0 + wave*16 + r + 8*hi;
      attn[((b*S_ + q)*H_ + h)*D_ + ot*16 + nl] = oacc[ot][r] / rsum[r];
    }
}

// ---------------------------------------------------------------------------
// Kernel 2: output projection  (B*S, 512) @ (512, 512) -> (B*S, 512)
// grid = (B*S/128, E/64), block = 256 threads (8 waves), 128x64 tile,
// K-chunk 64. LDS: As 16K + Bs 8K = 24 KB.
// ---------------------------------------------------------------------------
__global__ __launch_bounds__(256)
void out_proj_kernel(const float* __restrict__ A,  // (B*S, HD)
                     const float* __restrict__ W,  // (HD, E)
                     float* __restrict__ out)      // (B*S, E)
{
  __shared__ __align__(32) _Float16 As[128*64];  // [row][k]
  __shared__ __align__(32) _Float16 Bs[64*64];   // [n][k]

  const int r0 = blockIdx.x * 128, c0 = blockIdx.y * 64;
  const int tid  = threadIdx.x;
  const int wave = tid >> 5, lane = tid & 31;
  const int hi   = lane >> 4, nl = lane & 15;

  const v8f zero8 = {0.f,0.f,0.f,0.f,0.f,0.f,0.f,0.f};
  v8f cacc[4];
  #pragma unroll
  for (int nt = 0; nt < 4; ++nt) cacc[nt] = zero8;

  const int wk = tid >> 2, wnoff = (tid & 3) * 16;  // 64x64 W tile

  for (int kk = 0; kk < HD_/64; ++kk) {
    __syncthreads();
    // A tile: 2 chunks of 16 contiguous k each (all-b128 path)
    #pragma unroll
    for (int c = 0; c < 2; ++c) {
      int idx = tid + c*256;
      int row = idx >> 2, koff = (idx & 3) * 16;
      stage16(&As[row*64 + koff], &A[(r0 + row)*HD_ + kk*64 + koff]);
    }
    // W tile: coalesced along n, scattered into [n][k] LDS layout
    stage16_t(&Bs[wnoff*64 + wk], 64, &W[(kk*64 + wk)*E_ + c0 + wnoff]);
    __syncthreads();
    #pragma unroll
    for (int kk2 = 0; kk2 < 2; ++kk2) {
      v16h a = load_a16(&As[(wave*16 + nl)*64 + kk2*32 + hi*16]);
      #pragma unroll
      for (int nt = 0; nt < 4; ++nt) {
        v16h bm = load_b16(&Bs[(nt*16 + nl)*64 + kk2*32 + hi*16]);
        cacc[nt] = wmma16(a, bm, cacc[nt]);
      }
    }
  }

  #pragma unroll
  for (int nt = 0; nt < 4; ++nt)
    #pragma unroll
    for (int r = 0; r < 8; ++r)
      out[(r0 + wave*16 + r + 8*hi)*E_ + c0 + nt*16 + nl] = cacc[nt][r];
}

// ---------------------------------------------------------------------------
extern "C" void kernel_launch(void* const* d_in, const int* in_sizes, int n_in,
                              void* d_out, int out_size, void* d_ws, size_t ws_size,
                              hipStream_t stream) {
  const float*   X    = (const float*)d_in[0];   // inputs (B,S,E)
  const float*   Kg   = (const float*)d_in[1];   // keys (B,S,H,D)
  const float*   Vg   = (const float*)d_in[2];   // values (B,S,H,D)
  const float*   Wq   = (const float*)d_in[3];   // (E, H*D)
  const float*   Wo   = (const float*)d_in[4];   // (H*D, E)
  const uint8_t* mask = (const uint8_t*)d_in[5]; // (B,S,S) bool
  float* out  = (float*)d_out;                   // (B,S,E)
  float* attn = (float*)d_ws;                    // (B,S,H,D) f32 = 8 MB

  dim3 g1(S_/128, B_*H_);
  attn_fused_kernel<<<g1, 256, 0, stream>>>(X, Kg, Vg, Wq, mask, attn);

  dim3 g2((B_*S_)/128, E_/64);
  out_proj_kernel<<<g2, 256, 0, stream>>>(attn, Wo, out);
}
